// GlobalCrossAttention_56169582297632
// MI455X (gfx1250) — matrix operand order repaired and checked by reading, not compiled, not run on hardware
//
#include <hip/hip_runtime.h>
#include <hip/hip_bf16.h>
#include <math.h>

typedef __attribute__((ext_vector_type(16))) __bf16 v16bf;
typedef __attribute__((ext_vector_type(8)))  float  v8f;

#define BB 2
#define NQ 900
#define CC 256
#define HH 8
#define DHD 32
#define NN 4096
#define WIDW 64
#define RPE_HID 512

union BF16Tile { unsigned int u[8]; __bf16 h[16]; v16bf v; };

// ---------------------------------------------------------------------------
// Kernel 1: pack fp32 weight [256,256] (in,out) into tile-major bf16 pairs.
// Dword index layout: PW[(g*256 + n)*8 + j] where kp = (g>>1)*16 + (j>=4?8:0) + (g&1)*4 + (j&3) holds the pair
// (bf16(W[2kp,n]), bf16(W[2kp+1,n])).  A lane's 8 B-operand dwords for one
// WMMA step (g = ks*2 + hi) are contiguous -> two global_load_b128.
// ---------------------------------------------------------------------------
__global__ void pack_weights(const float* __restrict__ W, __bf16* __restrict__ P) {
  int idx = blockIdx.x * blockDim.x + threadIdx.x;
  if (idx >= 128 * 256) return;
  int j = idx & 7, n = (idx >> 3) & 255, g = idx >> 11;
  int kp = (g >> 1) * 16 + ((j >= 4) ? 8 : 0) + (g & 1) * 4 + (j & 3);
  P[2 * idx]     = (__bf16)W[(2 * kp) * 256 + n];
  P[2 * idx + 1] = (__bf16)W[(2 * kp + 1) * 256 + n];
}

// ---------------------------------------------------------------------------
// Kernel 2: projection GEMM (fp32 X [Mrows,256] @ packed bf16 W [256,256] + b)
// One wave computes a 16x16 tile, K=256 in 8 WMMA steps of 32.
// mode 0: q -> [b,h,nq,dh] bf16, scaled
// mode 1: k -> [b,h,n,dh]  bf16
// mode 2: v -> [b,h,dh,n]  bf16 (transposed for PV B-matrix loads)
// ---------------------------------------------------------------------------
__global__ void __launch_bounds__(32) proj_gemm(
    const float* __restrict__ X, const unsigned int* __restrict__ Wp,
    const float* __restrict__ bias, __bf16* __restrict__ out,
    int Mrows, int mode, float scale) {
  int lane = threadIdx.x;
  int hi = lane >> 4, ln = lane & 15;
  int m0 = blockIdx.x * 16, n0 = blockIdx.y * 16;
  int n = n0 + ln;
  int arow = min(m0 + ln, Mrows - 1);
  const float* xr = X + (size_t)arow * CC;

  v8f acc = {};
#pragma unroll
  for (int ks = 0; ks < 8; ++ks) {
    BF16Tile A, B;
    const float4* c1 = (const float4*)(xr + ks * 32 + hi * 8);
    float4 f0 = c1[0], f1 = c1[1];
    const float4* c2 = (const float4*)(xr + ks * 32 + 16 + hi * 8);
    float4 f2 = c2[0], f3 = c2[1];
    A.h[0]  = (__bf16)f0.x; A.h[1]  = (__bf16)f0.y;
    A.h[2]  = (__bf16)f0.z; A.h[3]  = (__bf16)f0.w;
    A.h[4]  = (__bf16)f1.x; A.h[5]  = (__bf16)f1.y;
    A.h[6]  = (__bf16)f1.z; A.h[7]  = (__bf16)f1.w;
    A.h[8]  = (__bf16)f2.x; A.h[9]  = (__bf16)f2.y;
    A.h[10] = (__bf16)f2.z; A.h[11] = (__bf16)f2.w;
    A.h[12] = (__bf16)f3.x; A.h[13] = (__bf16)f3.y;
    A.h[14] = (__bf16)f3.z; A.h[15] = (__bf16)f3.w;
    const uint4* bp4 = (const uint4*)(Wp + (size_t)((ks * 2 + hi) * 256 + n) * 8);
    uint4 b0 = bp4[0], b1 = bp4[1];
    B.u[0] = b0.x; B.u[1] = b0.y; B.u[2] = b0.z; B.u[3] = b0.w;
    B.u[4] = b1.x; B.u[5] = b1.y; B.u[6] = b1.z; B.u[7] = b1.w;
    acc = __builtin_amdgcn_wmma_f32_16x16x32_bf16(false, A.v, false, B.v,
                                                  (short)0, acc, false, false);
  }
  float bn = bias[n];
  int h = n >> 5, dh = n & 31;
#pragma unroll
  for (int j = 0; j < 8; ++j) {
    int m = m0 + j + 8 * hi;
    if (m >= Mrows) continue;
    __bf16 bv = (__bf16)((acc[j] + bn) * scale);
    if (mode == 0) {
      int b = m / NQ, qr = m % NQ;
      out[(((size_t)(b * HH + h) * NQ) + qr) * DHD + dh] = bv;
    } else if (mode == 1) {
      int b = m >> 12, key = m & (NN - 1);
      out[(((size_t)(b * HH + h) * NN) + key) * DHD + dh] = bv;
    } else {
      int b = m >> 12, key = m & (NN - 1);
      out[(((size_t)(b * HH + h) * DHD) + dh) * NN + key] = bv;
    }
  }
}

// ---------------------------------------------------------------------------
// Kernel 3: RPE MLP  relu(d @ W1 + b1) @ W2  per (b,q,col,axis)
// Weights in LDS; one wave per 8 grid columns; outputs [b,h,q,col] fp32.
// ---------------------------------------------------------------------------
__global__ void __launch_bounds__(256) rpe_mlp(
    const float* __restrict__ ref2d,
    const float* __restrict__ W1x, const float* __restrict__ b1x, const float* __restrict__ W2x,
    const float* __restrict__ W1y, const float* __restrict__ b1y, const float* __restrict__ W2y,
    float* __restrict__ rx, float* __restrict__ ry) {
  __shared__ __align__(16) float sW1[2 * RPE_HID];
  __shared__ __align__(16) float sB1[RPE_HID];
  __shared__ __align__(16) float sW2[RPE_HID * HH];
  int axis = blockIdx.y;
  const float* W1 = axis ? W1y : W1x;
  const float* b1 = axis ? b1y : b1x;
  const float* W2 = axis ? W2y : W2x;
  float* outp = axis ? ry : rx;
  int tid = threadIdx.x;
  for (int i = tid; i < 2 * RPE_HID; i += 256) sW1[i] = W1[i];
  for (int i = tid; i < RPE_HID; i += 256) sB1[i] = b1[i];
  for (int i = tid; i < RPE_HID * HH; i += 256) sW2[i] = W2[i];
  __syncthreads();

  int bq = blockIdx.x;
  int b = bq / NQ, q = bq - b * NQ;
  const float* r = ref2d + (size_t)bq * 4;
  float ctr = axis ? r[1] : r[0];
  float ext = axis ? r[3] : r[2];
  float v1 = ctr - ext * 0.5f, v2 = ctr + ext * 0.5f;
  int wave = tid >> 5, lane = tid & 31;

  for (int rr = 0; rr < 8; ++rr) {
    int col = wave * 8 + rr;
    float pos = ((float)col + 0.5f) * 16.0f;
    float d0 = v1 - pos, d1 = v2 - pos;
    float acc[8];
#pragma unroll
    for (int e = 0; e < 8; ++e) acc[e] = 0.f;
    int base = lane * 16;
#pragma unroll
    for (int t = 0; t < 4; ++t) {
      int i0 = base + t * 4;
      float4 wa = *(const float4*)&sW1[i0];
      float4 wb = *(const float4*)&sW1[RPE_HID + i0];
      float4 bb = *(const float4*)&sB1[i0];
      float he[4];
      he[0] = fmaxf(0.f, d0 * wa.x + d1 * wb.x + bb.x);
      he[1] = fmaxf(0.f, d0 * wa.y + d1 * wb.y + bb.y);
      he[2] = fmaxf(0.f, d0 * wa.z + d1 * wb.z + bb.z);
      he[3] = fmaxf(0.f, d0 * wa.w + d1 * wb.w + bb.w);
#pragma unroll
      for (int e = 0; e < 4; ++e) {
        float4 w2lo = *(const float4*)&sW2[(i0 + e) * 8];
        float4 w2hi = *(const float4*)&sW2[(i0 + e) * 8 + 4];
        acc[0] += he[e] * w2lo.x; acc[1] += he[e] * w2lo.y;
        acc[2] += he[e] * w2lo.z; acc[3] += he[e] * w2lo.w;
        acc[4] += he[e] * w2hi.x; acc[5] += he[e] * w2hi.y;
        acc[6] += he[e] * w2hi.z; acc[7] += he[e] * w2hi.w;
      }
    }
#pragma unroll
    for (int e = 0; e < 8; ++e) {
      float v = acc[e];
      v += __shfl_xor(v, 16, 32);
      v += __shfl_xor(v, 8, 32);
      v += __shfl_xor(v, 4, 32);
      v += __shfl_xor(v, 2, 32);
      v += __shfl_xor(v, 1, 32);
      acc[e] = v;
    }
    if (lane == 0) {
#pragma unroll
      for (int e = 0; e < 8; ++e)
        outp[(((size_t)(b * HH + e) * NQ) + q) * WIDW + col] = acc[e];
    }
  }
}

// ---------------------------------------------------------------------------
// Kernel 4: fused flash attention with decomposed RPE bias.
// One wave per (b,h, 16-query tile); 128 key blocks of 32.
// ---------------------------------------------------------------------------
__global__ void __launch_bounds__(32) flash_attn(
    const __bf16* __restrict__ qws, const __bf16* __restrict__ kws,
    const __bf16* __restrict__ vtws,
    const float* __restrict__ rx, const float* __restrict__ ry,
    const unsigned char* __restrict__ mask, __bf16* __restrict__ xout) {
  __shared__ __align__(16) float srx[16 * 64];
  __shared__ __align__(16) float sry[16 * 64];
  __shared__ __align__(16) float sp[16 * 32];
  __shared__ unsigned char smk[NN];

  int lane = threadIdx.x;
  int hi = lane >> 4, ln = lane & 15;
  int qt = blockIdx.x, bh = blockIdx.y;
  int b = bh >> 3, h = bh & 7;
  int q0 = qt * 16;

  for (int i = lane; i < 1024; i += 32) {
    int ql = i >> 6, col = i & 63;
    int q = min(q0 + ql, NQ - 1);
    srx[i] = rx[((size_t)bh * NQ + q) * WIDW + col];
    sry[i] = ry[((size_t)bh * NQ + q) * WIDW + col];
  }
  for (int i = lane; i < NN; i += 32) smk[i] = mask[(size_t)b * NN + i];
  __syncthreads();

  BF16Tile qA;
  {
    int q = min(q0 + ln, NQ - 1);
    const __bf16* qr = qws + ((size_t)bh * NQ + q) * DHD;
    uint4 a = *(const uint4*)(qr + hi * 16);
    uint4 c = *(const uint4*)(qr + hi * 16 + 8);
    qA.u[0] = a.x; qA.u[1] = a.y; qA.u[2] = a.z; qA.u[3] = a.w;
    qA.u[4] = c.x; qA.u[5] = c.y; qA.u[6] = c.z; qA.u[7] = c.w;
  }

  v8f Ol = {}, Or = {};
  float mrow[8], lrow[8];
#pragma unroll
  for (int j = 0; j < 8; ++j) { mrow[j] = -1e30f; lrow[j] = 0.f; }

  const __bf16* kbase = kws + (size_t)bh * NN * DHD;
  const __bf16* vbase = vtws + (size_t)bh * DHD * NN;

  for (int kb = 0; kb < 128; ++kb) {
    // Prefetch next block's K rows and V rows into cache (global_prefetch_b8).
    if (kb < 127) {
      __builtin_prefetch(kbase + (size_t)((kb + 1) * 32 + ln) * DHD + hi * 16, 0, 3);
      __builtin_prefetch(vbase + (size_t)ln * NN + (kb + 1) * 32 + hi * 16, 0, 3);
    }
    BF16Tile kB0, kB1;
    {
      const __bf16* kr = kbase + (size_t)(kb * 32 + ln) * DHD + hi * 16;
      uint4 a = *(const uint4*)(kr);
      uint4 c = *(const uint4*)(kr + 8);
      kB0.u[0] = a.x; kB0.u[1] = a.y; kB0.u[2] = a.z; kB0.u[3] = a.w;
      kB0.u[4] = c.x; kB0.u[5] = c.y; kB0.u[6] = c.z; kB0.u[7] = c.w;
    }
    {
      const __bf16* kr = kbase + (size_t)(kb * 32 + 16 + ln) * DHD + hi * 16;
      uint4 a = *(const uint4*)(kr);
      uint4 c = *(const uint4*)(kr + 8);
      kB1.u[0] = a.x; kB1.u[1] = a.y; kB1.u[2] = a.z; kB1.u[3] = a.w;
      kB1.u[4] = c.x; kB1.u[5] = c.y; kB1.u[6] = c.z; kB1.u[7] = c.w;
    }
    v8f S0 = {}, S1 = {};
    S0 = __builtin_amdgcn_wmma_f32_16x16x32_bf16(false, qA.v, false, kB0.v, (short)0, S0, false, false);
    S1 = __builtin_amdgcn_wmma_f32_16x16x32_bf16(false, qA.v, false, kB1.v, (short)0, S1, false, false);

    int rowg = kb >> 1;
    int c0 = (kb & 1) * 32;
    float mf0 = smk[kb * 32 + ln] ? -100.f : 0.f;
    float mf1 = smk[kb * 32 + 16 + ln] ? -100.f : 0.f;

#pragma unroll
    for (int j = 0; j < 8; ++j) {
      int mr = j + 8 * hi;
      float by = sry[mr * 64 + rowg];
      float s0 = S0[j] + by + srx[mr * 64 + c0 + ln] + mf0;
      float s1 = S1[j] + by + srx[mr * 64 + c0 + 16 + ln] + mf1;
      float t = fmaxf(s0, s1);
      t = fmaxf(t, __shfl_xor(t, 1, 32));
      t = fmaxf(t, __shfl_xor(t, 2, 32));
      t = fmaxf(t, __shfl_xor(t, 4, 32));
      t = fmaxf(t, __shfl_xor(t, 8, 32));
      float mnew = fmaxf(mrow[j], t);
      float alpha = __expf(mrow[j] - mnew);
      float p0 = __expf(s0 - mnew);
      float p1 = __expf(s1 - mnew);
      float rs = p0 + p1;
      rs += __shfl_xor(rs, 1, 32);
      rs += __shfl_xor(rs, 2, 32);
      rs += __shfl_xor(rs, 4, 32);
      rs += __shfl_xor(rs, 8, 32);
      lrow[j] = lrow[j] * alpha + rs;
      mrow[j] = mnew;
      Ol[j] *= alpha;
      Or[j] *= alpha;
      sp[mr * 32 + ln] = p0;
      sp[mr * 32 + 16 + ln] = p1;
    }
    __syncthreads();
    BF16Tile pA;
#pragma unroll
    for (int j = 0; j < 8; ++j) {
      pA.h[2 * j]     = (__bf16)sp[ln * 32 + hi * 16 + 2 * j];
      pA.h[2 * j + 1] = (__bf16)sp[ln * 32 + hi * 16 + 2 * j + 1];
    }
    BF16Tile vB0, vB1;
    {
      const __bf16* vr = vbase + (size_t)ln * NN + kb * 32 + hi * 16;
      uint4 a = *(const uint4*)(vr);
      uint4 c = *(const uint4*)(vr + 8);
      vB0.u[0] = a.x; vB0.u[1] = a.y; vB0.u[2] = a.z; vB0.u[3] = a.w;
      vB0.u[4] = c.x; vB0.u[5] = c.y; vB0.u[6] = c.z; vB0.u[7] = c.w;
    }
    {
      const __bf16* vr = vbase + (size_t)(16 + ln) * NN + kb * 32 + hi * 16;
      uint4 a = *(const uint4*)(vr);
      uint4 c = *(const uint4*)(vr + 8);
      vB1.u[0] = a.x; vB1.u[1] = a.y; vB1.u[2] = a.z; vB1.u[3] = a.w;
      vB1.u[4] = c.x; vB1.u[5] = c.y; vB1.u[6] = c.z; vB1.u[7] = c.w;
    }
    Ol = __builtin_amdgcn_wmma_f32_16x16x32_bf16(false, pA.v, false, vB0.v, (short)0, Ol, false, false);
    Or = __builtin_amdgcn_wmma_f32_16x16x32_bf16(false, pA.v, false, vB1.v, (short)0, Or, false, false);
    __syncthreads();
  }

#pragma unroll
  for (int j = 0; j < 8; ++j) {
    int q = q0 + j + 8 * hi;
    if (q >= NQ) continue;
    float inv = 1.0f / lrow[j];
    __bf16* o = xout + ((size_t)b * NQ + q) * CC + h * DHD;
    o[ln] = (__bf16)(Ol[j] * inv);
    o[16 + ln] = (__bf16)(Or[j] * inv);
  }
}

// ---------------------------------------------------------------------------
// Kernel 5: output projection  (bf16 X [1800,256] @ packed bf16 Wp + bp) -> f32
// ---------------------------------------------------------------------------
__global__ void __launch_bounds__(32) out_gemm(
    const __bf16* __restrict__ X, const unsigned int* __restrict__ Wp,
    const float* __restrict__ bias, float* __restrict__ out, int Mrows) {
  int lane = threadIdx.x;
  int hi = lane >> 4, ln = lane & 15;
  int m0 = blockIdx.x * 16, n0 = blockIdx.y * 16;
  int n = n0 + ln;
  int arow = min(m0 + ln, Mrows - 1);
  const __bf16* xr = X + (size_t)arow * CC;

  v8f acc = {};
#pragma unroll
  for (int ks = 0; ks < 8; ++ks) {
    BF16Tile A, B;
    uint4 a = *(const uint4*)(xr + ks * 32 + hi * 8);
    uint4 c = *(const uint4*)(xr + ks * 32 + 16 + hi * 8);
    A.u[0] = a.x; A.u[1] = a.y; A.u[2] = a.z; A.u[3] = a.w;
    A.u[4] = c.x; A.u[5] = c.y; A.u[6] = c.z; A.u[7] = c.w;
    const uint4* bp4 = (const uint4*)(Wp + (size_t)((ks * 2 + hi) * 256 + n) * 8);
    uint4 b0 = bp4[0], b1 = bp4[1];
    B.u[0] = b0.x; B.u[1] = b0.y; B.u[2] = b0.z; B.u[3] = b0.w;
    B.u[4] = b1.x; B.u[5] = b1.y; B.u[6] = b1.z; B.u[7] = b1.w;
    acc = __builtin_amdgcn_wmma_f32_16x16x32_bf16(false, A.v, false, B.v,
                                                  (short)0, acc, false, false);
  }
  float bn = bias[n];
#pragma unroll
  for (int j = 0; j < 8; ++j) {
    int m = m0 + j + 8 * hi;
    if (m >= Mrows) continue;
    out[(size_t)m * CC + n] = acc[j] + bn;
  }
}

// ---------------------------------------------------------------------------
extern "C" void kernel_launch(void* const* d_in, const int* in_sizes, int n_in,
                              void* d_out, int out_size, void* d_ws, size_t ws_size,
                              hipStream_t stream) {
  (void)in_sizes; (void)n_in; (void)out_size; (void)ws_size;
  const float* query = (const float*)d_in[1];
  const float* ref2d = (const float*)d_in[2];
  const float* kin   = (const float*)d_in[3];
  const float* vin   = (const float*)d_in[4];
  const unsigned char* mask = (const unsigned char*)d_in[6];
  const float* W1x = (const float*)d_in[7];
  const float* b1x = (const float*)d_in[8];
  const float* W2x = (const float*)d_in[9];
  const float* W1y = (const float*)d_in[10];
  const float* b1y = (const float*)d_in[11];
  const float* W2y = (const float*)d_in[12];
  const float* Wq = (const float*)d_in[13];
  const float* bq = (const float*)d_in[14];
  const float* Wk = (const float*)d_in[15];
  const float* bk = (const float*)d_in[16];
  const float* Wv = (const float*)d_in[17];
  const float* bv = (const float*)d_in[18];
  const float* Wpm = (const float*)d_in[19];
  const float* bp = (const float*)d_in[20];

  char* ws = (char*)d_ws;
  // workspace layout (all offsets 256B aligned)
  __bf16* WqP = (__bf16*)(ws + 0);            // 128*256*4 = 131072 B each
  __bf16* WkP = (__bf16*)(ws + 131072);
  __bf16* WvP = (__bf16*)(ws + 262144);
  __bf16* WpP = (__bf16*)(ws + 393216);
  __bf16* q_ws  = (__bf16*)(ws + 524288);   // 2*8*900*32*2  = 921600
  __bf16* k_ws  = (__bf16*)(ws + 1445888);  // 2*8*4096*32*2 = 4194304
  __bf16* vT_ws = (__bf16*)(ws + 5640192);  // 4194304
  float* rpe_x = (float*)(ws + 9834496);    // 2*8*900*64*4  = 3686400
  float* rpe_y = (float*)(ws + 13520896);   // 3686400
  __bf16* xattn = (__bf16*)(ws + 17207296); // 921600  -> 18128896 total

  const float scale = 0.1767766952966369f; // 32^-0.5

  pack_weights<<<128, 256, 0, stream>>>(Wq, WqP);
  pack_weights<<<128, 256, 0, stream>>>(Wk, WkP);
  pack_weights<<<128, 256, 0, stream>>>(Wv, WvP);
  pack_weights<<<128, 256, 0, stream>>>(Wpm, WpP);

  proj_gemm<<<dim3(113, 16), 32, 0, stream>>>(query, (const unsigned int*)WqP, bq, q_ws, BB * NQ, 0, scale);
  proj_gemm<<<dim3(512, 16), 32, 0, stream>>>(kin, (const unsigned int*)WkP, bk, k_ws, BB * NN, 1, 1.0f);
  proj_gemm<<<dim3(512, 16), 32, 0, stream>>>(vin, (const unsigned int*)WvP, bv, vT_ws, BB * NN, 2, 1.0f);

  rpe_mlp<<<dim3(BB * NQ, 2), 256, 0, stream>>>(ref2d, W1x, b1x, W2x, W1y, b1y, W2y,
                                                rpe_x, rpe_y);

  flash_attn<<<dim3(57, BB * HH), 32, 0, stream>>>(q_ws, k_ws, vT_ws, rpe_x, rpe_y,
                                                   mask, xattn);

  out_gemm<<<dim3(113, 16), 32, 0, stream>>>(xattn, (const unsigned int*)WpP, bp, (float*)d_out, BB * NQ);
}
